// MalwareDetector_52458730553357
// MI455X (gfx1250) — compile-verified, matrix operand order checked by reading, not run on hardware
//
#include <hip/hip_runtime.h>

typedef __attribute__((ext_vector_type(16))) _Float16 v16h;
typedef __attribute__((ext_vector_type(8)))  float    v8f;
typedef __attribute__((ext_vector_type(4)))  unsigned int uint4v;
typedef __attribute__((ext_vector_type(8)))  int      int8v;
typedef __attribute__((ext_vector_type(4)))  int      int4v;

#define N_NODES   100000
#define N_EDGES   1600000
#define N_GRAPHS  64
#define IN_DIM    512
#define LRELU     0.01f

// ---------------------------------------------------------------- utilities
__global__ void fill_zero_kernel(float* __restrict__ p, long long n) {
    long long i = (long long)blockIdx.x * blockDim.x + threadIdx.x;
    if (i < n) p[i] = 0.0f;
}

// Pack f32 weights [K,N] -> f16 in WMMA B-fragment order:
//   Wp[nt][c][lane][j],  nt = n/16 tile, c = k/32 chunk, lane = 32 lanes, j = 16 halves
//   element = W[c*32 + (lane>>4)*8 + (j<8 ? j : 8+j)][nt*16 + (lane&15)]
__global__ void pack_weights_kernel(const float* __restrict__ W,
                                    _Float16* __restrict__ Wp, int K, int N) {
    int idx = blockIdx.x * blockDim.x + threadIdx.x;
    if (idx >= K * N) return;
    const int slab = K * 16;          // halves per ntile slab
    const int nt = idx / slab;
    const int r  = idx % slab;
    const int c  = r / 512;           // 32 lanes * 16 halves per k-chunk
    const int q  = r % 512;
    const int L  = q / 16;
    const int j  = q % 16;
    const int n  = nt * 16 + (L & 15);
    const int k  = c * 32 + ((L >> 4) * 8) + (j < 8 ? j : 8 + j); // {0..7}U{16..23} striping
    Wp[idx] = (_Float16)W[k * N + n];
}

// ---------------------------------------------------------------- degrees / norms
__global__ void degree_kernel(const int* __restrict__ src, const int* __restrict__ dst,
                              float* __restrict__ deg_out, float* __restrict__ deg_in,
                              int n_edges) {
    int i = blockIdx.x * blockDim.x + threadIdx.x;
    if (i < n_edges) {
        atomicAdd(&deg_out[src[i]], 1.0f);
        atomicAdd(&deg_in [dst[i]], 1.0f);
    }
}

__global__ void norm_finalize_kernel(float* __restrict__ d, int n) {
    int i = blockIdx.x * blockDim.x + threadIdx.x;
    if (i < n) d[i] = rsqrtf(fmaxf(d[i], 1.0f));
}

// ---------------------------------------------------------------- WMMA GEMM
// out[M,N] = (A[M,K] * norm[M]) @ W[K,N], f16 operands / f32 accumulate.
// Whole weight matrix (pre-swizzled to B-fragment order) is TDM-staged into
// LDS once per block; each wave computes a full 16xN strip so every A
// fragment feeds N/16 independent WMMAs (no accumulator RAW chains).
template<int K, int N>
__global__ void gcn_gemm_wmma(const float* __restrict__ A,
                              const float* __restrict__ norm,
                              const _Float16* __restrict__ Wp,
                              float* __restrict__ out, int M) {
    constexpr int NT = N / 16;                    // ntile count (8 / 4 / 2)
    constexpr int CH = K / 32;                    // k-chunk count
    __shared__ __attribute__((aligned(32))) _Float16 wtile[K * N];

    // ---- TDM: stage swizzled weights (K*N halves) -> LDS, wave 0 issues ----
    if (threadIdx.x < 32) {
        unsigned long long ga = (unsigned long long)(uintptr_t)Wp;
        unsigned lds = (unsigned)(uintptr_t)&wtile[0];   // low 32 bits = LDS offset
        // D# group 0: count=1 | lds_addr | global_addr[56:0] | type=2
        uint4v g0;
        g0[0] = 1u;
        g0[1] = lds;
        g0[2] = (unsigned)(ga & 0xffffffffu);
        g0[3] = (unsigned)((ga >> 32) & 0x01ffffffu) | (2u << 30);
        // D# group 1: data_size=1(2B); tensor_dim0=K*16; tensor_dim1=NT;
        //             tile_dim0=K*16; tile_dim1=NT; tensor_dim0_stride=K*16
        constexpr unsigned TD0 = (unsigned)(K * 16);      // <= 8192, fits 16 bits
        int8v g1;
        g1[0] = (int)(1u << 16);                                  // data_size
        g1[1] = (int)((TD0 & 0xffffu) << 16);                     // tensor_dim0 lo
        g1[2] = (int)(((TD0 >> 16) & 0xffffu)
                      | (((unsigned)NT & 0xffffu) << 16));        // td0 hi | td1 lo
        g1[3] = (int)((((unsigned)NT >> 16) & 0xffffu)
                      | ((TD0 & 0xffffu) << 16));                 // td1 hi | tile_dim0
        g1[4] = (int)((unsigned)NT & 0xffffu);                    // tile_dim1 (tile_dim2=0)
        g1[5] = (int)TD0;                                         // tensor_dim0_stride lo32
        g1[6] = 0;
        g1[7] = 0;
        int4v gz4 = {0, 0, 0, 0};
        int8v gz8 = {0, 0, 0, 0, 0, 0, 0, 0};
        __builtin_amdgcn_tensor_load_to_lds(g0, g1, gz4, gz4, gz8, 0);
        __builtin_amdgcn_s_wait_tensorcnt(0);     // issuing wave waits for TDM
    }
    __syncthreads();                               // publish LDS tile to all waves

    const int lane  = threadIdx.x & 31;
    const int wave  = threadIdx.x >> 5;
    const int mtile = (blockIdx.x * 8 + wave) * 16;
    if (mtile >= M) return;                        // wave-uniform (EXEC all-1s)

    const int l16 = lane & 15;
    const int hi  = lane >> 4;                     // 0 or 1
    const int kb  = hi * 8;                        // K sub-chunk base within 32

    const int   mr = mtile + l16;                  // A row this lane feeds
    const float s  = norm[mr];
    const float* __restrict__ arow = A + (size_t)mr * K;

    v8f acc[NT] = {};
    #pragma unroll 2
    for (int c = 0; c < CH; ++c) {
        v16h a;
        // A fragment: halves 0..7 = K[c*32+kb ..), 8..15 = K[c*32+kb+16 ..)
        #pragma unroll
        for (int i = 0; i < 8; ++i)
            a[i] = (_Float16)(arow[c * 32 + kb + i] * s);
        #pragma unroll
        for (int i = 0; i < 8; ++i)
            a[8 + i] = (_Float16)(arow[c * 32 + kb + 16 + i] * s);
        // one A fragment feeds NT independent WMMAs
        #pragma unroll
        for (int nt = 0; nt < NT; ++nt) {
            const v16h b = *(const v16h*)&wtile[nt * (K * 16) + c * 512 + lane * 16];
            acc[nt] = __builtin_amdgcn_wmma_f32_16x16x32_f16(
                false, a, false, b, (short)0, acc[nt], false, false);
        }
    }

    // D layout: vgpr r holds row (mtile + r + 8*hi), column nt*16 + l16
    const int mbase = mtile + hi * 8;
    #pragma unroll
    for (int nt = 0; nt < NT; ++nt) {
        const int nc = nt * 16 + l16;
        #pragma unroll
        for (int r = 0; r < 8; ++r)
            out[(size_t)(mbase + r) * N + nc] = acc[nt][r];
    }
}

// ---------------------------------------------------------------- edge scatter-add
__global__ void edge_scatter_kernel(const float* __restrict__ proj,
                                    const int* __restrict__ src,
                                    const int* __restrict__ dst,
                                    float* __restrict__ accum,
                                    int n_edges, int dim) {
    const int chunk = dim >> 2;
    long long idx = (long long)blockIdx.x * blockDim.x + threadIdx.x;
    if (idx >= (long long)n_edges * chunk) return;
    const int e = (int)(idx / chunk);
    const int c = (int)(idx % chunk);
    const int s = src[e], d = dst[e];
    const float4 v = ((const float4*)(proj + (size_t)s * dim))[c];
    float* p = accum + (size_t)d * dim + (size_t)c * 4;
    atomicAdd(p + 0, v.x);
    atomicAdd(p + 1, v.y);
    atomicAdd(p + 2, v.z);
    atomicAdd(p + 3, v.w);
}

// ---------------------------------------------------------------- norm_dst * leaky_relu
__global__ void post_norm_lrelu_kernel(float* __restrict__ accum,
                                       const float* __restrict__ norm_dst,
                                       long long total, int dim) {
    long long i = (long long)blockIdx.x * blockDim.x + threadIdx.x;
    if (i >= total) return;
    const int node = (int)(i / dim);
    float v = accum[i] * norm_dst[node];
    accum[i] = v > 0.0f ? v : LRELU * v;
}

// ---------------------------------------------------------------- graph pooling
__global__ void pool_cnt_kernel(const int* __restrict__ gid, float* __restrict__ cnt, int n) {
    int i = blockIdx.x * blockDim.x + threadIdx.x;
    if (i < n) atomicAdd(&cnt[gid[i]], 1.0f);
}

__global__ void pool_sum_kernel(const float* __restrict__ h, const int* __restrict__ gid,
                                float* __restrict__ hg, int n_nodes) {
    long long i = (long long)blockIdx.x * blockDim.x + threadIdx.x;
    if (i >= (long long)n_nodes * 32) return;
    const int node = (int)(i >> 5);
    const int f    = (int)(i & 31);
    atomicAdd(&hg[gid[node] * 32 + f], h[i]);
}

// ---------------------------------------------------------------- classifier head
__global__ void final_head_kernel(const float* __restrict__ hg, const float* __restrict__ cnt,
                                  const float* __restrict__ Wc, const float* __restrict__ bc,
                                  float* __restrict__ out) {
    int t = threadIdx.x;
    if (t >= N_GRAPHS * 5) return;
    const int g = t / 5, o = t % 5;
    const float inv = 1.0f / fmaxf(cnt[g], 1.0f);
    float acc = bc[o];
    #pragma unroll
    for (int f = 0; f < 32; ++f)
        acc += (hg[g * 32 + f] * inv) * Wc[f * 5 + o];
    out[t] = acc;
}

// ---------------------------------------------------------------- launcher
extern "C" void kernel_launch(void* const* d_in, const int* in_sizes, int n_in,
                              void* d_out, int out_size, void* d_ws, size_t ws_size,
                              hipStream_t stream) {
    const float* features = (const float*)d_in[0];
    const int*   src      = (const int*)  d_in[1];
    const int*   dst      = (const int*)  d_in[2];
    const int*   gid      = (const int*)  d_in[3];
    const float* W1       = (const float*)d_in[4];
    const float* W2       = (const float*)d_in[5];
    const float* W3       = (const float*)d_in[6];
    const float* Wc       = (const float*)d_in[7];
    const float* bc       = (const float*)d_in[8];
    float*       out      = (float*)d_out;

    const int M = N_NODES;
    const int E = N_EDGES;

    // Workspace layout (floats)
    float* wsf      = (float*)d_ws;
    float* norm_src = wsf;                        // 100000
    float* norm_dst = wsf + 100000;               // 100000
    float* hg       = wsf + 200000;               // 2048 (64x32)
    float* cnt      = wsf + 202048;               // 64
    float* bufA     = wsf + 202112;               // 100000 x 128
    float* bufB     = bufA + (size_t)M * 128;     // 100000 x 128
    _Float16* whbase = (_Float16*)(bufB + (size_t)M * 128);
    _Float16* Wp1 = whbase;                       // 512*128 (swizzled)
    _Float16* Wp2 = Wp1 + 512 * 128;              // 128*64
    _Float16* Wp3 = Wp2 + 128 * 64;               // 64*32

    const int T = 256;

    // 1) degrees -> norms (reuse norm buffers as degree accumulators)
    fill_zero_kernel<<<(200000 + T - 1) / T, T, 0, stream>>>(norm_src, 200000);
    degree_kernel<<<(E + T - 1) / T, T, 0, stream>>>(src, dst, norm_src, norm_dst, E);
    norm_finalize_kernel<<<(200000 + T - 1) / T, T, 0, stream>>>(norm_src, 200000);

    // 2) weights -> f16, pre-swizzled into WMMA B-fragment order
    pack_weights_kernel<<<(512 * 128 + T - 1) / T, T, 0, stream>>>(W1, Wp1, 512, 128);
    pack_weights_kernel<<<(128 * 64  + T - 1) / T, T, 0, stream>>>(W2, Wp2, 128, 64);
    pack_weights_kernel<<<(64  * 32  + T - 1) / T, T, 0, stream>>>(W3, Wp3, 64, 32);

    // 3) three GraphConv layers
    const int mtiles = M / 16;                    // 6250 exactly
    const int gx     = (mtiles + 7) / 8;          // 8 waves per block

    const int Nd[3] = {128, 64, 32};
    for (int L = 0; L < 3; ++L) {
        const int Nl = Nd[L];
        // projection: bufA = (h * norm_src) @ W   [TDM-staged B, WMMA math]
        if (L == 0) {
            gcn_gemm_wmma<512, 128><<<gx, T, 0, stream>>>(features, norm_src, Wp1, bufA, M);
        } else if (L == 1) {
            gcn_gemm_wmma<128, 64><<<gx, T, 0, stream>>>(bufB, norm_src, Wp2, bufA, M);
        } else {
            gcn_gemm_wmma<64, 32><<<gx, T, 0, stream>>>(bufB, norm_src, Wp3, bufA, M);
        }
        // scatter-add over edges into bufB
        const long long tot = (long long)M * Nl;
        fill_zero_kernel<<<(int)((tot + T - 1) / T), T, 0, stream>>>(bufB, tot);
        const long long sthreads = (long long)E * (Nl >> 2);
        edge_scatter_kernel<<<(int)((sthreads + T - 1) / T), T, 0, stream>>>(
            bufA, src, dst, bufB, E, Nl);
        // norm_dst scaling + leaky_relu, in place (bufB becomes next layer input)
        post_norm_lrelu_kernel<<<(int)((tot + T - 1) / T), T, 0, stream>>>(
            bufB, norm_dst, tot, Nl);
    }

    // 4) mean-pool per graph + classifier head
    fill_zero_kernel<<<(2048 + 64 + T - 1) / T, T, 0, stream>>>(hg, 2048 + 64);
    pool_cnt_kernel<<<(M + T - 1) / T, T, 0, stream>>>(gid, cnt, M);
    pool_sum_kernel<<<(int)(((long long)M * 32 + T - 1) / T), T, 0, stream>>>(bufB, gid, hg, M);
    final_head_kernel<<<1, 320, 0, stream>>>(hg, cnt, Wc, bc, out);
}
